// GNNAutoencoder_31842887532756
// MI455X (gfx1250) — matrix operand order repaired and checked by reading, not compiled, not run on hardware
//
#include <hip/hip_runtime.h>
#include <hip/hip_bf16.h>

// ---------------------------------------------------------------------------
// GNN autoencoder (DGCNN-like) for gfx1250.
// EdgeConv layers run on v_wmma_f32_16x16x32_f16: one wave handles NPW=2
// nodes; each node's 16 contiguous incoming edges form an M=16 tile. The two
// nodes share every B (weight) fragment and provide independent WMMA chains
// to hide the WMMA RAW hazard slots.
// ---------------------------------------------------------------------------

typedef _Float16 v16h __attribute__((ext_vector_type(16)));
typedef _Float16 v8h  __attribute__((ext_vector_type(8)));
typedef _Float16 v4h  __attribute__((ext_vector_type(4)));
typedef float    v8f  __attribute__((ext_vector_type(8)));

#define GN_N 32768
#define GN_K 16
#define GN_B 32
#define GN_P 1024

// ---------------------------------------------------------------------------
// Repack an fp32 weight matrix W[Kdim, Nout] into f16 WMMA B-fragments.
// Fragment layout (per 16x16 N-tile, per K-step of 32):
//   lane L holds column (L&15); its 16 contiguous halfs j=0..15 are
//   K rows (kstep*32 + (L>>4)*8 + (j&7) + (j>>3)*16).  K rows >= Kdim are zero padding.
// Linear index: (((ntile*ksteps + kstep)*32 + lane)*16 + j)
// ---------------------------------------------------------------------------
__global__ void pack_weights_kernel(const float* __restrict__ W,
                                    _Float16* __restrict__ out,
                                    int Kdim, int Kpad, int Nout) {
    int idx = blockIdx.x * blockDim.x + threadIdx.x;
    int ksteps = Kpad >> 5;
    int total = (Nout >> 4) * ksteps * 512;
    if (idx >= total) return;
    int j     = idx & 15;
    int lane  = (idx >> 4) & 31;
    int rest  = idx >> 9;
    int kstep = rest % ksteps;
    int ntile = rest / ksteps;
    int col   = (ntile << 4) + (lane & 15);
    int krow  = (kstep << 5) + ((lane >> 4) << 3) + (j & 7) + ((j >> 3) << 4);
    float v = (krow < Kdim) ? W[(size_t)krow * Nout + col] : 0.0f;
    out[idx] = (_Float16)v;
}

// ---------------------------------------------------------------------------
// EdgeConv: one wave per NPW nodes.
//   A  = [x_i | x_j - x_i]            (16 x 2*F_IN, zero-padded to K2)
//   H  = relu(A @ Wa + ba)            (16 x F_OUT)
//   D  = H @ Wb + bb                  (16 x F_OUT)
//   Y[i] = max over the 16 rows of D  (segment_max; dst is sorted, K=16)
// ---------------------------------------------------------------------------
template <int F_IN, int F_OUT, int K2, int WAVES, int NPW>
__global__ __launch_bounds__(WAVES * 32)
void edgeconv_kernel(const float* __restrict__ X,
                     const int* __restrict__ src,
                     const _Float16* __restrict__ WaP, const float* __restrict__ ba,
                     const _Float16* __restrict__ WbP, const float* __restrict__ bb,
                     float* __restrict__ Y) {
    constexpr int AP  = K2 + 8;        // LDS pitch in halfs (multiple of 8)
    constexpr int HP  = F_OUT + 8;
    constexpr int KS1 = K2 / 32;       // K-steps of GEMM1
    constexpr int KS2 = F_OUT / 32;    // K-steps of GEMM2
    constexpr int NT  = F_OUT / 16;    // N-tiles

    __shared__ __align__(16) _Float16 Ash[WAVES][NPW][16 * AP];
    __shared__ __align__(16) _Float16 Hsh[WAVES][NPW][16 * HP];

    const int lane  = threadIdx.x & 31;
    const int wave  = threadIdx.x >> 5;
    const int node0 = (blockIdx.x * WAVES + wave) * NPW;

    // ---- Phase 1: build A tiles (rows = each node's 16 edges) ----
    {
        const int row = lane & 15;
        const int h   = lane >> 4;       // lane pair: which half of columns
        #pragma unroll
        for (int n = 0; n < NPW; ++n) {
            const int node = node0 + n;
            _Float16* A = Ash[wave][n];
            const int s = src[node * GN_K + row];
            const float* xi = X + (size_t)node * F_IN;
            const float* xj = X + (size_t)s * F_IN;

            if constexpr (F_IN % 8 == 0) {
                const int cbase = h * (F_IN / 2);
                #pragma unroll
                for (int c0 = 0; c0 < F_IN / 2; c0 += 4) {
                    const int c = cbase + c0;
                    float4 vi = *(const float4*)(xi + c);
                    float4 vj = *(const float4*)(xj + c);
                    v4h ai = { (_Float16)vi.x, (_Float16)vi.y,
                               (_Float16)vi.z, (_Float16)vi.w };
                    v4h ad = { (_Float16)(vj.x - vi.x), (_Float16)(vj.y - vi.y),
                               (_Float16)(vj.z - vi.z), (_Float16)(vj.w - vi.w) };
                    *(v4h*)(A + row * AP + c)        = ai;
                    *(v4h*)(A + row * AP + F_IN + c) = ad;
                }
            } else {
                for (int c = h; c < F_IN; c += 2) {
                    float vi = xi[c];
                    float vj = xj[c];
                    A[row * AP + c]        = (_Float16)vi;
                    A[row * AP + F_IN + c] = (_Float16)(vj - vi);
                }
            }
            if constexpr (2 * F_IN < K2) {
                for (int c = 2 * F_IN + h; c < K2; c += 2)
                    A[row * AP + c] = (_Float16)0.f;
            }
        }
    }
    __syncthreads();

    const int row   = lane & 15;
    const int kb    = (lane >> 4) << 3;   // 0 or 8 (A-fragment K offset)
    const int col   = lane & 15;
    const int rbase = (lane >> 4) << 3;   // D rows r+rbase

    // ---- Phase 2: H = relu(A @ Wa + ba) ----
    for (int nt = 0; nt < NT; ++nt) {
        v8f acc[NPW];
        #pragma unroll
        for (int n = 0; n < NPW; ++n) acc[n] = (v8f){};

        for (int ks = 0; ks < KS1; ++ks) {
            v16h b = *(const v16h*)(WaP + ((size_t)(nt * KS1 + ks) * 32 + lane) * 16);
            #pragma unroll
            for (int n = 0; n < NPW; ++n) {
                const _Float16* ap = Ash[wave][n] + row * AP + ks * 32 + kb;
                v8h lo = *(const v8h*)ap;
                v8h hi = *(const v8h*)(ap + 16);
                v16h a;
                #pragma unroll
                for (int j = 0; j < 8; ++j) { a[j] = lo[j]; a[8 + j] = hi[j]; }
                acc[n] = __builtin_amdgcn_wmma_f32_16x16x32_f16(
                            false, a, false, b, (short)0, acc[n], false, false);
            }
        }
        float bav = ba[nt * 16 + col];
        #pragma unroll
        for (int n = 0; n < NPW; ++n) {
            _Float16* H = Hsh[wave][n];
            #pragma unroll
            for (int r = 0; r < 8; ++r) {
                float v = acc[n][r] + bav;
                v = v > 0.f ? v : 0.f;
                H[(r + rbase) * HP + nt * 16 + col] = (_Float16)v;
            }
        }
    }
    __syncthreads();

    // ---- Phase 3: D = H @ Wb + bb, then per-column max over 16 rows ----
    for (int nt = 0; nt < NT; ++nt) {
        v8f acc[NPW];
        #pragma unroll
        for (int n = 0; n < NPW; ++n) acc[n] = (v8f){};

        for (int ks = 0; ks < KS2; ++ks) {
            v16h b = *(const v16h*)(WbP + ((size_t)(nt * KS2 + ks) * 32 + lane) * 16);
            #pragma unroll
            for (int n = 0; n < NPW; ++n) {
                const _Float16* hp = Hsh[wave][n] + row * HP + ks * 32 + kb;
                v8h lo = *(const v8h*)hp;
                v8h hi = *(const v8h*)(hp + 16);
                v16h a;
                #pragma unroll
                for (int j = 0; j < 8; ++j) { a[j] = lo[j]; a[8 + j] = hi[j]; }
                acc[n] = __builtin_amdgcn_wmma_f32_16x16x32_f16(
                            false, a, false, b, (short)0, acc[n], false, false);
            }
        }
        float bbv = bb[nt * 16 + col];
        #pragma unroll
        for (int n = 0; n < NPW; ++n) {
            float m = acc[n][0] + bbv;
            #pragma unroll
            for (int r = 1; r < 8; ++r) m = fmaxf(m, acc[n][r] + bbv);
            m = fmaxf(m, __shfl_xor(m, 16, 32));   // combine row halves
            if (lane < 16)
                Y[(size_t)(node0 + n) * F_OUT + nt * 16 + col] = m;
        }
    }
}

// ---------------------------------------------------------------------------
// Graph max-pool: [N,256] -> [B,256]  (1024 sorted nodes per graph)
// ---------------------------------------------------------------------------
__global__ void pool_kernel(const float* __restrict__ X3, float* __restrict__ pool) {
    int b = blockIdx.x;
    int c = threadIdx.x;
    float m = -3.4e38f;
    const float* base = X3 + (size_t)b * GN_P * 256 + c;
    for (int p = 0; p < GN_P; ++p)
        m = fmaxf(m, base[(size_t)p * 256]);
    pool[b * 256 + c] = m;
}

// ---------------------------------------------------------------------------
// Small dense head in fp32 (negligible FLOPs): out = [relu](in @ W + b)
// ---------------------------------------------------------------------------
__global__ void fc_kernel(const float* __restrict__ in, const float* __restrict__ W,
                          const float* __restrict__ bias, float* __restrict__ out,
                          int Kdim, int Nout, int rows, int do_relu) {
    int idx = blockIdx.x * blockDim.x + threadIdx.x;
    if (idx >= rows * Nout) return;
    int r = idx / Nout;
    int o = idx % Nout;
    float s = bias[o];
    const float* inr = in + (size_t)r * Kdim;
    for (int k = 0; k < Kdim; ++k)
        s += inr[k] * W[(size_t)k * Nout + o];
    if (do_relu) s = s > 0.f ? s : 0.f;
    out[idx] = s;
}

// ---------------------------------------------------------------------------
extern "C" void kernel_launch(void* const* d_in, const int* in_sizes, int n_in,
                              void* d_out, int out_size, void* d_ws, size_t ws_size,
                              hipStream_t stream) {
    const float* x    = (const float*)d_in[0];
    const int*   eidx = (const int*)d_in[1];        // [2,E]: first E = src
    const float* W1a  = (const float*)d_in[3];
    const float* b1a  = (const float*)d_in[4];
    const float* W1b  = (const float*)d_in[5];
    const float* b1b  = (const float*)d_in[6];
    const float* W2a  = (const float*)d_in[7];
    const float* b2a  = (const float*)d_in[8];
    const float* W2b  = (const float*)d_in[9];
    const float* b2b  = (const float*)d_in[10];
    const float* W3a  = (const float*)d_in[11];
    const float* b3a  = (const float*)d_in[12];
    const float* W3b  = (const float*)d_in[13];
    const float* b3b  = (const float*)d_in[14];
    const float* Wfc  = (const float*)d_in[15];
    const float* bfc  = (const float*)d_in[16];
    const float* Wd1  = (const float*)d_in[17];
    const float* bd1  = (const float*)d_in[18];
    const float* Wd2  = (const float*)d_in[19];
    const float* bd2  = (const float*)d_in[20];

    const int* src = eidx;  // edge_index[0]

    // Workspace carve-out (256B aligned slices)
    char* p = (char*)d_ws;
    auto alloc = [&](size_t bytes) -> void* {
        void* r = (void*)p;
        p += (bytes + 255) & ~(size_t)255;
        return r;
    };
    _Float16* pW1a = (_Float16*)alloc((size_t)32  * 64  * 2);
    _Float16* pW1b = (_Float16*)alloc((size_t)64  * 64  * 2);
    _Float16* pW2a = (_Float16*)alloc((size_t)128 * 128 * 2);
    _Float16* pW2b = (_Float16*)alloc((size_t)128 * 128 * 2);
    _Float16* pW3a = (_Float16*)alloc((size_t)256 * 256 * 2);
    _Float16* pW3b = (_Float16*)alloc((size_t)256 * 256 * 2);
    float* x1   = (float*)alloc((size_t)GN_N * 64  * 4);
    float* x2   = (float*)alloc((size_t)GN_N * 128 * 4);
    float* x3   = (float*)alloc((size_t)GN_N * 256 * 4);
    float* pool = (float*)alloc((size_t)GN_B * 256 * 4);
    float* zb   = (float*)alloc((size_t)GN_B * 128 * 4);
    float* hb   = (float*)alloc((size_t)GN_B * 256 * 4);

    // --- Repack weights into WMMA B-fragment layout (f16) ---
    auto packN = [](int Kpad, int Nout) { return (Nout >> 4) * (Kpad >> 5) * 512; };
    {
        int t;
        t = packN(32, 64);
        pack_weights_kernel<<<(t + 255) / 256, 256, 0, stream>>>(W1a, pW1a, 6, 32, 64);
        t = packN(64, 64);
        pack_weights_kernel<<<(t + 255) / 256, 256, 0, stream>>>(W1b, pW1b, 64, 64, 64);
        t = packN(128, 128);
        pack_weights_kernel<<<(t + 255) / 256, 256, 0, stream>>>(W2a, pW2a, 128, 128, 128);
        pack_weights_kernel<<<(t + 255) / 256, 256, 0, stream>>>(W2b, pW2b, 128, 128, 128);
        t = packN(256, 256);
        pack_weights_kernel<<<(t + 255) / 256, 256, 0, stream>>>(W3a, pW3a, 256, 256, 256);
        pack_weights_kernel<<<(t + 255) / 256, 256, 0, stream>>>(W3b, pW3b, 256, 256, 256);
    }

    // --- EdgeConv stack: NPW=2 nodes per wave ---
    {
        // Layers 1-2: 4 waves/block; layer 3: 2 waves/block (LDS budget).
        dim3 g12(GN_N / (4 * 2)), b12(4 * 32);
        edgeconv_kernel<3, 64, 32, 4, 2><<<g12, b12, 0, stream>>>(
            x, src, pW1a, b1a, pW1b, b1b, x1);
        edgeconv_kernel<64, 128, 128, 4, 2><<<g12, b12, 0, stream>>>(
            x1, src, pW2a, b2a, pW2b, b2b, x2);
        dim3 g3(GN_N / (2 * 2)), b3(2 * 32);
        edgeconv_kernel<128, 256, 256, 2, 2><<<g3, b3, 0, stream>>>(
            x2, src, pW3a, b3a, pW3b, b3b, x3);
    }

    // --- Pool + decoder head (fp32) ---
    pool_kernel<<<GN_B, 256, 0, stream>>>(x3, pool);
    {
        int t;
        t = GN_B * 128;
        fc_kernel<<<(t + 255) / 256, 256, 0, stream>>>(pool, Wfc, bfc, zb, 256, 128, GN_B, 0);
        t = GN_B * 256;
        fc_kernel<<<(t + 255) / 256, 256, 0, stream>>>(zb, Wd1, bd1, hb, 128, 256, GN_B, 1);
        t = GN_B * GN_P * 3;
        fc_kernel<<<(t + 255) / 256, 256, 0, stream>>>(hb, Wd2, bd2, (float*)d_out,
                                                       256, GN_P * 3, GN_B, 0);
    }
}
